// BidirectionalMambaBlock_12446815224153
// MI455X (gfx1250) — compile-verified, matrix operand order checked by reading, not run on hardware
//
#include <hip/hip_runtime.h>
#include <cstddef>

// ---------------------------------------------------------------------------
// Bidirectional Mamba block for gfx1250 (MI455X), bf16 WMMA pipeline with
// async global->LDS tile staging (double buffered) in the GEMMs.
// ---------------------------------------------------------------------------

#define D_MODEL 768
#define D_STATE 16
#define D_CONV  4
#define D_INNER 1536
#define DT_RANK 48
#define DT_PAD  64           // DT_RANK padded to WMMA K granularity
#define BATCH   2
#define SEQLEN  2048
#define NROWS   (BATCH * SEQLEN)   // 4096 tokens
#define XDBL_W  (DT_RANK + 2 * D_STATE)  // 80

typedef __bf16 v8bf  __attribute__((ext_vector_type(8)));
typedef __bf16 v16bf __attribute__((ext_vector_type(16)));
typedef float  v8f   __attribute__((ext_vector_type(8)));

#define MODE_BF16     0
#define MODE_F32      1
#define MODE_SOFTPLUS 2

// ---------------------------------------------------------------------------
// helpers
// ---------------------------------------------------------------------------

// 32-bit LDS byte address (addrspace(3)) of a __shared__ object, for use as
// the VDST operand of global_load_async_to_lds_*.
__device__ __forceinline__ unsigned lds_addr_of(const void* p) {
  return (unsigned)(unsigned long long)(
      __attribute__((address_space(3))) const char*)p;
}

// One 16-byte async global->LDS copy per lane (GLOBAL_LOAD_ASYNC_TO_LDS_B128,
// tracked by ASYNCcnt).
__device__ __forceinline__ void async_copy_b128(unsigned lds_byte_addr,
                                                const void* gptr) {
  asm volatile("global_load_async_to_lds_b128 %0, %1, off"
               :
               : "v"(lds_byte_addr), "v"((unsigned long long)(size_t)gptr)
               : "memory");
}

// Build one 16x32 bf16 WMMA fragment from an LDS row (stride-40 bf16 rows).
// Lane layout (CDNA5 ISA 7.12.2, 16-bit A 16x32): lane half selects the
// 8-wide K sub-runs {half*8 .. half*8+7} and {16+half*8 .. 16+half*8+7}.
__device__ __forceinline__ v16bf load_frag(const __bf16* rowbase, int half) {
  const v8bf lo = *reinterpret_cast<const v8bf*>(rowbase + half * 8);
  const v8bf hi = *reinterpret_cast<const v8bf*>(rowbase + 16 + half * 8);
  v16bf f;
#pragma unroll
  for (int i = 0; i < 8; ++i) { f[i] = lo[i]; f[8 + i] = hi[i]; }
  return f;
}

// ---------------------------------------------------------------------------
// 1) RMSNorm -> bf16 tokens
// ---------------------------------------------------------------------------
__global__ __launch_bounds__(256) void rmsnorm_kernel(
    const float* __restrict__ x, const float* __restrict__ w,
    __bf16* __restrict__ xn) {
  const int row = blockIdx.x;
  const int tid = threadIdx.x;
  __shared__ float red[256];
  float ss = 0.0f;
  for (int c = tid; c < D_MODEL; c += 256) {
    float v = x[(size_t)row * D_MODEL + c];
    ss += v * v;
  }
  red[tid] = ss;
  __syncthreads();
  for (int s = 128; s > 0; s >>= 1) {
    if (tid < s) red[tid] += red[tid + s];
    __syncthreads();
  }
  const float rs = rsqrtf(red[0] / (float)D_MODEL + 1e-5f);
  for (int c = tid; c < D_MODEL; c += 256) {
    xn[(size_t)row * D_MODEL + c] =
        (__bf16)(x[(size_t)row * D_MODEL + c] * rs * w[c]);
  }
}

// ---------------------------------------------------------------------------
// 2) weight conversion kernels
// ---------------------------------------------------------------------------
__global__ __launch_bounds__(256) void cvt_f32_bf16_kernel(
    const float* __restrict__ src, __bf16* __restrict__ dst, int n) {
  int i = blockIdx.x * 256 + threadIdx.x;
  if (i < n) dst[i] = (__bf16)src[i];
}

// dt_proj_w (1536 x 48) -> zero-padded (1536 x 64) bf16
__global__ __launch_bounds__(256) void pad_dtw_kernel(
    const float* __restrict__ src, __bf16* __restrict__ dst) {
  int i = blockIdx.x * 256 + threadIdx.x;  // over 1536*64
  if (i >= D_INNER * DT_PAD) return;
  int col = i & (DT_PAD - 1);
  int row = i >> 6;
  dst[i] = (col < DT_RANK) ? (__bf16)src[row * DT_RANK + col] : (__bf16)0.0f;
}

// ---------------------------------------------------------------------------
// 3) Generic bf16 GEMM: C[M,N] = A[M,K] * W[N,K]^T  (v_wmma_f32_16x16x32_bf16)
//    Block tile 64(M) x 128(N), 8 waves (4x2), K-step 32, f32 accumulate.
//    Tiles staged with async global->LDS copies, double buffered:
//    issue k+1 tile, s_wait_asynccnt 3 (current tile done), barrier, WMMA.
// ---------------------------------------------------------------------------

// Issue one K-tile's async copies (3 b128 instructions per thread/wave).
__device__ __forceinline__ void issue_tile(
    const __bf16* __restrict__ A, const __bf16* __restrict__ Bw, int M, int N,
    int K, int mb, int nb, int k0, int tid, __bf16* bufA, __bf16* bufB) {
  {
    const int idx = tid << 3;          // 8 bf16 per thread
    const int r = idx >> 5, cc = idx & 31;
    async_copy_b128(lds_addr_of(bufA + r * 40 + cc),
                    A + (size_t)(mb + r) * K + (k0 + cc));
  }
#pragma unroll
  for (int it = 0; it < 2; ++it) {
    const int idx = (tid + (it << 8)) << 3;
    const int r = idx >> 5, cc = idx & 31;
    int gr = nb + r;
    if (gr >= N) gr = N - 1;  // clamp: dead data, masked at epilogue
    async_copy_b128(lds_addr_of(bufB + r * 40 + cc),
                    Bw + (size_t)gr * K + (k0 + cc));
  }
}

__global__ __launch_bounds__(256) void gemm_bf16_wmma_kernel(
    const __bf16* __restrict__ A, const __bf16* __restrict__ Bw,
    float* __restrict__ Cf, __bf16* __restrict__ Cbf,
    const float* __restrict__ bias, int M, int N, int K, int mode) {
  __shared__ __bf16 ldsA[2][64 * 40];   // stride 40 bf16 (80B) vs. conflicts
  __shared__ __bf16 ldsB[2][128 * 40];

  const int tid  = threadIdx.x;
  const int lane = tid & 31;
  const int wid  = tid >> 5;
  const int wm   = wid & 3;    // wave row: 16 M-rows each
  const int wn   = wid >> 2;   // wave col: 64 N-cols each
  const int mb   = blockIdx.y * 64;
  const int nb   = blockIdx.x * 128;
  const int half = lane >> 4;
  const int l16  = lane & 15;

  v8f acc[4];
#pragma unroll
  for (int s = 0; s < 4; ++s)
#pragma unroll
    for (int j = 0; j < 8; ++j) acc[s][j] = 0.0f;

  // prologue: stage first tile into buffer 0
  issue_tile(A, Bw, M, N, K, mb, nb, 0, tid, ldsA[0], ldsB[0]);

  int buf = 0;
  for (int k0 = 0; k0 < K; k0 += 32, buf ^= 1) {
    const bool has_next = (k0 + 32) < K;
    if (has_next) {
      // overlap: stage next tile into the other buffer
      issue_tile(A, Bw, M, N, K, mb, nb, k0 + 32, tid, ldsA[buf ^ 1],
                 ldsB[buf ^ 1]);
      // ASYNCcnt completes in order: <=3 outstanding => current tile landed
      asm volatile("s_wait_asynccnt 0x3" ::: "memory");
      // prefetch tile k+2 A-rows into caches
      if (k0 + 64 < K) {
        const int idx = tid << 3;
        const int r = idx >> 5, cc = idx & 31;
        __builtin_prefetch(A + (size_t)(mb + r) * K + (k0 + 64 + cc), 0, 1);
      }
    } else {
      asm volatile("s_wait_asynccnt 0x0" ::: "memory");
    }
    __syncthreads();  // all waves' async writes visible

    const v16bf af = load_frag(&ldsA[buf][(wm * 16 + l16) * 40], half);
#pragma unroll
    for (int s = 0; s < 4; ++s) {
      const v16bf bf = load_frag(&ldsB[buf][(wn * 64 + s * 16 + l16) * 40],
                                 half);
      acc[s] = __builtin_amdgcn_wmma_f32_16x16x32_bf16(
          /*neg_a=*/false, af, /*neg_b=*/false, bf,
          /*c_mod=*/(short)0, acc[s], /*reuse_a=*/false, /*reuse_b=*/false);
    }
    __syncthreads();  // protect buffer reuse two iterations later
  }

  // ---- epilogue (C/D layout: VGPR j -> M = j + 8*half, N = lane&15) ----
#pragma unroll
  for (int s = 0; s < 4; ++s) {
    const int col = nb + wn * 64 + s * 16 + l16;
    if (col < N) {
#pragma unroll
      for (int j = 0; j < 8; ++j) {
        const int row = mb + wm * 16 + j + (half << 3);
        if (row < M) {
          float v = acc[s][j];
          const size_t o = (size_t)row * N + col;
          if (mode == MODE_SOFTPLUS) {
            v += bias[col];
            v = (v > 20.0f) ? v : log1pf(__expf(v));
            Cf[o] = v;
          } else if (mode == MODE_F32) {
            Cf[o] = v;
          } else {
            Cbf[o] = (__bf16)v;
          }
        }
      }
    }
  }
}

// ---------------------------------------------------------------------------
// 4) depthwise causal conv(4) + bias + SiLU (direction-aware, no data flips)
//    fwd: u[t] = silu(sum_j w[j]*xi[t-3+j] + b)
//    bwd: u[t] = silu(sum_j w[3-j]*xi[t+j] + b)   (== conv on flipped seq)
// ---------------------------------------------------------------------------
__global__ __launch_bounds__(256) void conv_silu_kernel(
    const __bf16* __restrict__ xz, const float* __restrict__ cw,
    const float* __restrict__ cb, __bf16* __restrict__ u, int dir) {
  const int gid = blockIdx.x * 256 + threadIdx.x;  // over NROWS*D_INNER
  const int c = gid % D_INNER;
  const int row = gid / D_INNER;
  const int b = row >> 11;
  const int t = row & (SEQLEN - 1);
  float acc = cb[c];
  if (dir == 0) {
#pragma unroll
    for (int j = 0; j < D_CONV; ++j) {
      const int tt = t - 3 + j;
      if (tt >= 0)
        acc += cw[c * D_CONV + j] *
               (float)xz[((size_t)(b * SEQLEN + tt)) * (2 * D_INNER) + c];
    }
  } else {
#pragma unroll
    for (int j = 0; j < D_CONV; ++j) {
      const int tt = t + j;
      if (tt < SEQLEN)
        acc += cw[c * D_CONV + 3 - j] *
               (float)xz[((size_t)(b * SEQLEN + tt)) * (2 * D_INNER) + c];
    }
  }
  const float s = acc / (1.0f + __expf(-acc));  // silu
  u[(size_t)row * D_INNER + c] = (__bf16)s;
}

// ---------------------------------------------------------------------------
// 5) pack x_dbl[:, :48] into zero-padded bf16 (rows x 64) for the dt GEMM
// ---------------------------------------------------------------------------
__global__ __launch_bounds__(256) void build_dtin_kernel(
    const float* __restrict__ xdbl, __bf16* __restrict__ dtin) {
  const int gid = blockIdx.x * 256 + threadIdx.x;  // over NROWS*64
  const int col = gid & (DT_PAD - 1);
  const int row = gid >> 6;
  dtin[gid] =
      (col < DT_RANK) ? (__bf16)xdbl[(size_t)row * XDBL_W + col] : (__bf16)0.0f;
}

// ---------------------------------------------------------------------------
// 6) selective scan: one lane per channel; h[16] in registers; B_t/C_t
//    broadcast through LDS. dir=1 iterates time in reverse (== flip/flip).
// ---------------------------------------------------------------------------
__global__ __launch_bounds__(256) void scan_kernel(
    const __bf16* __restrict__ u, const float* __restrict__ dt,
    const float* __restrict__ xdbl, const __bf16* __restrict__ xz,
    const float* __restrict__ A_log, const float* __restrict__ D_skip,
    __bf16* __restrict__ y, int dir) {
  const int tid = threadIdx.x;
  const int cblk = blockIdx.x % (D_INNER / 256);
  const int b = blockIdx.x / (D_INNER / 256);
  const int c = cblk * 256 + tid;

  float Ar[D_STATE];
#pragma unroll
  for (int n = 0; n < D_STATE; ++n) Ar[n] = -__expf(A_log[c * D_STATE + n]);
  const float Dv = D_skip[c];

  float h[D_STATE];
#pragma unroll
  for (int n = 0; n < D_STATE; ++n) h[n] = 0.0f;

  __shared__ float sB[D_STATE];
  __shared__ float sC[D_STATE];

  for (int s = 0; s < SEQLEN; ++s) {
    const int t = dir ? (SEQLEN - 1 - s) : s;
    const int row = b * SEQLEN + t;
    if (tid < D_STATE)
      sB[tid] = xdbl[(size_t)row * XDBL_W + DT_RANK + tid];
    else if (tid < 2 * D_STATE)
      sC[tid - D_STATE] = xdbl[(size_t)row * XDBL_W + DT_RANK + D_STATE +
                               (tid - D_STATE)];
    __syncthreads();

    const float dtv = dt[(size_t)row * D_INNER + c];
    const float uv = (float)u[(size_t)row * D_INNER + c];
    const float du = dtv * uv;
    float yv = 0.0f;
#pragma unroll
    for (int n = 0; n < D_STATE; ++n) {
      h[n] = __expf(dtv * Ar[n]) * h[n] + du * sB[n];
      yv += h[n] * sC[n];
    }
    yv += uv * Dv;
    const float zv = (float)xz[(size_t)row * (2 * D_INNER) + D_INNER + c];
    yv *= zv / (1.0f + __expf(-zv));  // * silu(z)
    y[(size_t)row * D_INNER + c] = (__bf16)yv;
    __syncthreads();
  }
}

// ---------------------------------------------------------------------------
// 7) final: out = out_fwd + out_bwd + residual
// ---------------------------------------------------------------------------
__global__ __launch_bounds__(256) void final_add_kernel(
    const float* __restrict__ op0, const float* __restrict__ op1,
    const float* __restrict__ x, float* __restrict__ out) {
  const int i = blockIdx.x * 256 + threadIdx.x;  // over NROWS*D_MODEL
  out[i] = op0[i] + op1[i] + x[i];
}

// ---------------------------------------------------------------------------
// host launcher
// ---------------------------------------------------------------------------
extern "C" void kernel_launch(void* const* d_in, const int* in_sizes, int n_in,
                              void* d_out, int out_size, void* d_ws,
                              size_t ws_size, hipStream_t stream) {
  (void)in_sizes; (void)n_in; (void)out_size; (void)ws_size;

  const float* x      = (const float*)d_in[0];
  const float* norm_w = (const float*)d_in[1];
  // params: (in_proj, conv_w, conv_b, x_proj, dt_w, dt_b, A_log, D_skip, out_proj)
  const float* P[2][9];
  for (int d = 0; d < 2; ++d)
    for (int i = 0; i < 9; ++i) P[d][i] = (const float*)d_in[2 + d * 9 + i];

  // ---- workspace layout (all offsets 256B-aligned) ----
  char* ws = (char*)d_ws;
  size_t off = 0;
  __bf16* xn = (__bf16*)(ws + off);                 off += (size_t)NROWS * D_MODEL * 2;
  __bf16* w_in = (__bf16*)(ws + off);               off += (size_t)2 * 2 * D_INNER * D_MODEL * 2;
  __bf16* w_xp = (__bf16*)(ws + off);               off += (size_t)2 * XDBL_W * D_INNER * 2;
  __bf16* w_dt = (__bf16*)(ws + off);               off += (size_t)2 * D_INNER * DT_PAD * 2;
  __bf16* w_out = (__bf16*)(ws + off);              off += (size_t)2 * D_MODEL * D_INNER * 2;
  __bf16* xz = (__bf16*)(ws + off);                 off += (size_t)2 * NROWS * 2 * D_INNER * 2;
  __bf16* u = (__bf16*)(ws + off);                  off += (size_t)2 * NROWS * D_INNER * 2;
  float* xdbl = (float*)(ws + off);                 off += (size_t)2 * NROWS * XDBL_W * 4;
  __bf16* dtin = (__bf16*)(ws + off);               off += (size_t)2 * NROWS * DT_PAD * 2;
  float* dtbuf = (float*)(ws + off);                off += (size_t)2 * NROWS * D_INNER * 4;
  __bf16* ybuf = (__bf16*)(ws + off);               off += (size_t)2 * NROWS * D_INNER * 2;
  float* opart = (float*)(ws + off);                off += (size_t)2 * NROWS * D_MODEL * 4;

  // ---- weight conversion (fp32 -> bf16, dt_w zero-padded to K=64) ----
  for (int d = 0; d < 2; ++d) {
    int n1 = 2 * D_INNER * D_MODEL;   // in_proj
    cvt_f32_bf16_kernel<<<(n1 + 255) / 256, 256, 0, stream>>>(
        P[d][0], w_in + (size_t)d * n1, n1);
    int n2 = XDBL_W * D_INNER;        // x_proj
    cvt_f32_bf16_kernel<<<(n2 + 255) / 256, 256, 0, stream>>>(
        P[d][3], w_xp + (size_t)d * n2, n2);
    int n3 = D_MODEL * D_INNER;       // out_proj
    cvt_f32_bf16_kernel<<<(n3 + 255) / 256, 256, 0, stream>>>(
        P[d][8], w_out + (size_t)d * n3, n3);
    pad_dtw_kernel<<<(D_INNER * DT_PAD + 255) / 256, 256, 0, stream>>>(
        P[d][4], w_dt + (size_t)d * D_INNER * DT_PAD);
  }

  // ---- RMSNorm ----
  rmsnorm_kernel<<<NROWS, 256, 0, stream>>>(x, norm_w, xn);

  for (int d = 0; d < 2; ++d) {
    __bf16* xz_d   = xz + (size_t)d * NROWS * 2 * D_INNER;
    __bf16* u_d    = u + (size_t)d * NROWS * D_INNER;
    float*  xdbl_d = xdbl + (size_t)d * NROWS * XDBL_W;
    __bf16* dtin_d = dtin + (size_t)d * NROWS * DT_PAD;
    float*  dt_d   = dtbuf + (size_t)d * NROWS * D_INNER;
    __bf16* y_d    = ybuf + (size_t)d * NROWS * D_INNER;
    float*  op_d   = opart + (size_t)d * NROWS * D_MODEL;

    // in_proj: (4096 x 3072) = xn(4096x768) @ W^T
    gemm_bf16_wmma_kernel<<<dim3((2 * D_INNER + 127) / 128, NROWS / 64), 256, 0,
                            stream>>>(
        xn, w_in + (size_t)d * 2 * D_INNER * D_MODEL, nullptr, xz_d, nullptr,
        NROWS, 2 * D_INNER, D_MODEL, MODE_BF16);

    // conv + SiLU
    conv_silu_kernel<<<(NROWS * D_INNER) / 256, 256, 0, stream>>>(
        xz_d, P[d][1], P[d][2], u_d, d);

    // x_proj: (4096 x 80) = u(4096x1536) @ W^T
    gemm_bf16_wmma_kernel<<<dim3((XDBL_W + 127) / 128, NROWS / 64), 256, 0,
                            stream>>>(
        u_d, w_xp + (size_t)d * XDBL_W * D_INNER, xdbl_d, nullptr, nullptr,
        NROWS, XDBL_W, D_INNER, MODE_F32);

    // dt input packing + dt projection (+bias, softplus fused)
    build_dtin_kernel<<<(NROWS * DT_PAD) / 256, 256, 0, stream>>>(xdbl_d,
                                                                  dtin_d);
    gemm_bf16_wmma_kernel<<<dim3((D_INNER + 127) / 128, NROWS / 64), 256, 0,
                            stream>>>(
        dtin_d, w_dt + (size_t)d * D_INNER * DT_PAD, dt_d, nullptr, P[d][5],
        NROWS, D_INNER, DT_PAD, MODE_SOFTPLUS);

    // selective scan (+ D skip + silu(z) gate), direction-aware
    scan_kernel<<<BATCH * (D_INNER / 256), 256, 0, stream>>>(
        u_d, dt_d, xdbl_d, xz_d, P[d][6], P[d][7], y_d, d);

    // out_proj: (4096 x 768) = y(4096x1536) @ W^T
    gemm_bf16_wmma_kernel<<<dim3((D_MODEL + 127) / 128, NROWS / 64), 256, 0,
                            stream>>>(
        y_d, w_out + (size_t)d * D_MODEL * D_INNER, op_d, nullptr, nullptr,
        NROWS, D_MODEL, D_INNER, MODE_F32);
  }

  // out = fwd + bwd + residual
  final_add_kernel<<<(NROWS * D_MODEL) / 256, 256, 0, stream>>>(
      opart, opart + (size_t)NROWS * D_MODEL, x, (float*)d_out);
}
